// TermEncoder_47940424958092
// MI455X (gfx1250) — matrix operand order repaired and checked by reading, not compiled, not run on hardware
//
#include <hip/hip_runtime.h>

typedef __attribute__((ext_vector_type(16))) _Float16 v16h;
typedef __attribute__((ext_vector_type(8)))  _Float16 v8h;
typedef __attribute__((ext_vector_type(4)))  _Float16 v4h;
typedef __attribute__((ext_vector_type(8)))  float    v8f;

#define NNODES  262144
#define NEDGES  262144
#define DIM     256
#define NGRAPHS 512

// ---------------- utility: zero a float buffer (float4 stores) ----------------
__global__ void zero_f4(float* __restrict__ p, int n4) {
  int i = blockIdx.x * blockDim.x + threadIdx.x;
  if (i < n4) ((float4*)p)[i] = make_float4(0.f, 0.f, 0.f, 0.f);
}

// ---------------- h16 = (f16) emb_table[node_types] ----------------
__global__ void embed_gather(const int* __restrict__ types,
                             const float* __restrict__ emb,
                             _Float16* __restrict__ h16) {
  int idx  = blockIdx.x * blockDim.x + threadIdx.x;   // NNODES * 64 threads
  int node = idx >> 6;
  int c    = (idx & 63) << 2;
  int t    = types[node];
  float4 v = *(const float4*)(emb + (size_t)t * DIM + c);
  v4h o;
  o[0] = (_Float16)v.x; o[1] = (_Float16)v.y;
  o[2] = (_Float16)v.z; o[3] = (_Float16)v.w;
  *(v4h*)(h16 + (size_t)node * DIM + c) = o;
}

// ---------------- Wt[hop][mat][n][k] = (f16) W[hop][k][n]  (B^T, f16) --------
__global__ void prep_weights(const float* __restrict__ Wself,
                             const float* __restrict__ Wup,
                             const float* __restrict__ Wdown,
                             _Float16* __restrict__ Wt) {
  int idx = blockIdx.x * blockDim.x + threadIdx.x;    // 2*3*256*256 threads
  int m   = idx >> 16;                                // hop*3 + mat
  int r   = idx & 65535;
  int n   = r >> 8, k = r & 255;
  int hop = m / 3, mat = m % 3;
  const float* src = (mat == 0) ? Wself : (mat == 1) ? Wup : Wdown;
  Wt[idx] = (_Float16)src[(size_t)hop * DIM * DIM + (size_t)k * DIM + n];
}

// -------- edge scatter: aggUp[dst]+=h[src]; aggDown[src]+=h[dst]  (f16->f32) -
__global__ void edge_scatter(const int* __restrict__ esrc,
                             const int* __restrict__ edst,
                             const _Float16* __restrict__ h16,
                             float* __restrict__ aggUp,
                             float* __restrict__ aggDown) {
  int idx = blockIdx.x * blockDim.x + threadIdx.x;    // NEDGES * 64 threads
  int e   = idx >> 6;
  int c   = (idx & 63) << 2;
  int s = esrc[e], d = edst[e];
  v4h hs = *(const v4h*)(h16 + (size_t)s * DIM + c);
  v4h hd = *(const v4h*)(h16 + (size_t)d * DIM + c);
  float* up = aggUp   + (size_t)d * DIM + c;
  float* dn = aggDown + (size_t)s * DIM + c;
  atomicAdd(up + 0, (float)hs[0]); atomicAdd(up + 1, (float)hs[1]);
  atomicAdd(up + 2, (float)hs[2]); atomicAdd(up + 3, (float)hs[3]);
  atomicAdd(dn + 0, (float)hd[0]); atomicAdd(dn + 1, (float)hd[1]);
  atomicAdd(dn + 2, (float)hd[2]); atomicAdd(dn + 3, (float)hd[3]);
}

// ---------------- fused hop GEMM: relu([h|up|down] @ Wt^T + b) ----------------
// Block = 256 threads (8 waves), tile = 64 rows x 256 cols.
// Wave w: M-subtile pair (w&1), N-group (w>>1): 2x4 = 8 WMMA tiles per wave.
// All 4 B fragments are loaded up front (one clause, incremental waits), then
// register-reused across both M-subtiles: 8 WMMAs per K-step.
// K = 768 in steps of 32 (v_wmma_f32_16x16x32_f16).
__global__ __launch_bounds__(256)
void fused_hop_gemm(const _Float16* __restrict__ h16,
                    const float* __restrict__ aggUp,
                    const float* __restrict__ aggDown,
                    const _Float16* __restrict__ Wt,   // [3][256][256] (n-major)
                    const float* __restrict__ bias,    // [256]
                    _Float16* __restrict__ hOut16) {
  const int lane = threadIdx.x & 31;
  const int wave = threadIdx.x >> 5;
  const int mp   = (wave & 1) * 32;           // M-subtile pair base (rows)
  const int ng   = wave >> 1;                 // N group 0..3 -> cols ng*64..+63
  const int l15  = lane & 15;
  const int lhi  = lane >> 4;
  const int mBase = blockIdx.x * 64 + mp;

  v8f acc[2][4] = {};

  const int rowA0  = mBase + l15;             // first M-subtile row for this lane
  const int khalfA = lhi * 8;                 // A-frag: K = khalfA..+7, 16+khalfA..+7
  const int kbB    = lhi * 8;                 // B-frag: K = kbB..+7, 16+kbB..+7

  for (int kk = 0; kk < 3 * DIM; kk += 32) {
    const int col = kk & (DIM - 1);

    // --- issue all 4 B-fragment loads first (L2-resident weights) ---
    const _Float16* wbase = Wt + ((size_t)(kk >> 8) * DIM * DIM) + col + kbB;
    v16h bf[4];
#pragma unroll
    for (int t = 0; t < 4; ++t) {
      int n = ng * 64 + t * 16 + l15;
      v8h blo = *(const v8h*)(wbase + (size_t)n * DIM);
      v8h bhi = *(const v8h*)(wbase + (size_t)n * DIM + 16);
#pragma unroll
      for (int j = 0; j < 8; ++j) { bf[t][j] = blo[j]; bf[t][8 + j] = bhi[j]; }
    }

    // --- build the two A fragments ---
    v16h a0, a1;
    if (kk < DIM) {
      // A from f16 h: direct fragment loads (2x b128 per frag)
      const _Float16* ap0 = h16 + (size_t)rowA0 * DIM + col + khalfA;
      const _Float16* ap1 = ap0 + (size_t)16 * DIM;
      v8h lo0 = *(const v8h*)(ap0);
      v8h hi0 = *(const v8h*)(ap0 + 16);
      v8h lo1 = *(const v8h*)(ap1);
      v8h hi1 = *(const v8h*)(ap1 + 16);
      __builtin_prefetch(ap0 + 32, 0, 1);
#pragma unroll
      for (int j = 0; j < 8; ++j) {
        a0[j] = lo0[j]; a0[8 + j] = hi0[j];
        a1[j] = lo1[j]; a1[8 + j] = hi1[j];
      }
    } else {
      // A from f32 edge aggregates: load + on-the-fly convert
      const float* Asrc = (kk < 2 * DIM) ? aggUp : aggDown;
      const float* ap0 = Asrc + (size_t)rowA0 * DIM + col + khalfA;
      const float* ap1 = ap0 + (size_t)16 * DIM;
      float af0[16], af1[16];
      *(float4*)(af0 + 0)  = *(const float4*)(ap0);
      *(float4*)(af0 + 4)  = *(const float4*)(ap0 + 4);
      *(float4*)(af0 + 8)  = *(const float4*)(ap0 + 16);
      *(float4*)(af0 + 12) = *(const float4*)(ap0 + 20);
      *(float4*)(af1 + 0)  = *(const float4*)(ap1);
      *(float4*)(af1 + 4)  = *(const float4*)(ap1 + 4);
      *(float4*)(af1 + 8)  = *(const float4*)(ap1 + 16);
      *(float4*)(af1 + 12) = *(const float4*)(ap1 + 20);
      __builtin_prefetch(ap0 + 32, 0, 1);
#pragma unroll
      for (int j = 0; j < 16; ++j) {
        a0[j] = (_Float16)af0[j];
        a1[j] = (_Float16)af1[j];
      }
    }

    // --- 8 WMMAs; B fragments consumed in load order so waits drain gradually
#pragma unroll
    for (int t = 0; t < 4; ++t) {
      acc[0][t] = __builtin_amdgcn_wmma_f32_16x16x32_f16(
          false, a0, false, bf[t], (short)0, acc[0][t], false, false);
      acc[1][t] = __builtin_amdgcn_wmma_f32_16x16x32_f16(
          false, a1, false, bf[t], (short)0, acc[1][t], false, false);
    }
  }

  // Epilogue: bias + relu, f16 out. D-layout: VGPR v -> row lhi*8+v, col l15.
#pragma unroll
  for (int u = 0; u < 2; ++u) {
#pragma unroll
    for (int t = 0; t < 4; ++t) {
      int coln = ng * 64 + t * 16 + l15;
      float bv = bias[coln];
#pragma unroll
      for (int v = 0; v < 8; ++v) {
        int row = mBase + u * 16 + lhi * 8 + v;
        float val = fmaxf(acc[u][t][v] + bv, 0.0f);
        hOut16[(size_t)row * DIM + coln] = (_Float16)val;
      }
    }
  }
}

// ---------------- pooling: sorted graph_ids, run-length + boundary atomics ---
__global__ void pool_sum(const int* __restrict__ gids,
                         const _Float16* __restrict__ h16,
                         float* __restrict__ out) {
  int node0 = blockIdx.x * 32;                // NNODES/32 blocks, 64 threads
  int c = threadIdx.x << 2;
  float4 accv = make_float4(0.f, 0.f, 0.f, 0.f);
  int curg = gids[node0];
  for (int i = 0; i < 32; ++i) {
    int node = node0 + i;
    int g = gids[node];
    if (g != curg) {
      float* o = out + (size_t)curg * DIM + c;
      atomicAdd(o + 0, accv.x); atomicAdd(o + 1, accv.y);
      atomicAdd(o + 2, accv.z); atomicAdd(o + 3, accv.w);
      accv = make_float4(0.f, 0.f, 0.f, 0.f);
      curg = g;
    }
    v4h v = *(const v4h*)(h16 + (size_t)node * DIM + c);
    accv.x += (float)v[0]; accv.y += (float)v[1];
    accv.z += (float)v[2]; accv.w += (float)v[3];
  }
  float* o = out + (size_t)curg * DIM + c;
  atomicAdd(o + 0, accv.x); atomicAdd(o + 1, accv.y);
  atomicAdd(o + 2, accv.z); atomicAdd(o + 3, accv.w);
}

__global__ void graph_counts(const int* __restrict__ gids,
                             float* __restrict__ counts) {
  int g = blockIdx.x * blockDim.x + threadIdx.x;
  if (g >= NGRAPHS) return;
  int lo0 = 0, hi0 = NNODES;                  // lower_bound(g)
  while (lo0 < hi0) { int mid = (lo0 + hi0) >> 1; if (gids[mid] < g) lo0 = mid + 1; else hi0 = mid; }
  int lo1 = 0, hi1 = NNODES;                  // lower_bound(g+1)
  while (lo1 < hi1) { int mid = (lo1 + hi1) >> 1; if (gids[mid] < g + 1) lo1 = mid + 1; else hi1 = mid; }
  counts[g] = (float)(lo1 - lo0);
}

__global__ void finalize_mean(float* __restrict__ out,
                              const float* __restrict__ counts) {
  int idx = blockIdx.x * blockDim.x + threadIdx.x;  // NGRAPHS*64 threads
  int g = idx >> 6;
  int c = (idx & 63) << 2;
  float inv = 1.0f / fmaxf(counts[g], 1.0f);
  float4* o = (float4*)(out + (size_t)g * DIM + c);
  float4 v = *o;
  v.x *= inv; v.y *= inv; v.z *= inv; v.w *= inv;
  *o = v;
}

// -----------------------------------------------------------------------------
extern "C" void kernel_launch(void* const* d_in, const int* in_sizes, int n_in,
                              void* d_out, int out_size, void* d_ws, size_t ws_size,
                              hipStream_t stream) {
  const int*   node_types = (const int*)d_in[0];
  const int*   edge_src   = (const int*)d_in[1];
  const int*   edge_dst   = (const int*)d_in[2];
  const int*   graph_ids  = (const int*)d_in[3];
  const float* emb        = (const float*)d_in[4];
  const float* W_up       = (const float*)d_in[5];
  const float* W_down     = (const float*)d_in[6];
  const float* W_self     = (const float*)d_in[7];
  const float* bias       = (const float*)d_in[8];
  float* out = (float*)d_out;

  const size_t HD32 = (size_t)NNODES * DIM * sizeof(float);
  const size_t HD16 = (size_t)NNODES * DIM * sizeof(_Float16);
  char* ws = (char*)d_ws;
  size_t off = 0;
  _Float16*  hA      = (_Float16*)(ws + off); off += HD16;
  _Float16*  hB      = (_Float16*)(ws + off); off += HD16;
  float*     aggUp   = (float*)(ws + off); off += HD32;
  float*     aggDown = (float*)(ws + off); off += HD32;
  _Float16*  Wt      = (_Float16*)(ws + off); off += (size_t)2 * 3 * DIM * DIM * sizeof(_Float16);
  float*     counts  = (float*)(ws + off); off += NGRAPHS * sizeof(float);

  const int n4agg   = NNODES * DIM / 4;               // 16M float4
  const int aggBlks = n4agg / 256;
  const int n4out   = NGRAPHS * DIM / 4;

  // init
  zero_f4<<<aggBlks, 256, 0, stream>>>(aggUp, n4agg);
  zero_f4<<<aggBlks, 256, 0, stream>>>(aggDown, n4agg);
  zero_f4<<<(n4out + 255) / 256, 256, 0, stream>>>(out, n4out);
  embed_gather<<<NNODES * 64 / 256, 256, 0, stream>>>(node_types, emb, hA);
  prep_weights<<<2 * 3 * DIM * DIM / 256, 256, 0, stream>>>(W_self, W_up, W_down, Wt);

  // hop 0
  edge_scatter<<<NEDGES * 64 / 256, 256, 0, stream>>>(edge_src, edge_dst, hA, aggUp, aggDown);
  fused_hop_gemm<<<NNODES / 64, 256, 0, stream>>>(hA, aggUp, aggDown, Wt, bias, hB);

  // hop 1
  zero_f4<<<aggBlks, 256, 0, stream>>>(aggUp, n4agg);
  zero_f4<<<aggBlks, 256, 0, stream>>>(aggDown, n4agg);
  edge_scatter<<<NEDGES * 64 / 256, 256, 0, stream>>>(edge_src, edge_dst, hB, aggUp, aggDown);
  fused_hop_gemm<<<NNODES / 64, 256, 0, stream>>>(hB, aggUp, aggDown,
                                                  Wt + (size_t)3 * DIM * DIM, bias + DIM, hA);

  // pooling
  graph_counts<<<2, 256, 0, stream>>>(graph_ids, counts);
  pool_sum<<<NNODES / 32, 64, 0, stream>>>(graph_ids, hA, out);
  finalize_mean<<<NGRAPHS * 64 / 256, 256, 0, stream>>>(out, counts);
}
